// InstantNGP_40415642255460
// MI455X (gfx1250) — matrix operand order repaired and checked by reading, not compiled, not run on hardware
//
#include <hip/hip_runtime.h>
#include <hip/hip_bf16.h>

typedef __attribute__((ext_vector_type(16))) _Float16 v16h;
typedef __attribute__((ext_vector_type(8)))  _Float16 v8h;
typedef __attribute__((ext_vector_type(8)))  float    v8f;

#define LVL 16
#define TBL 524288            // T = 2^19 entries per level, F=2 floats each
#define BLK 128               // threads per block = 4 waves = 128 points
#define ENC_K 64              // padded feature dim (41 -> 64)

// floor(16 * 1.611^l), l = 0..15 (matches reference RES)
__device__ __constant__ int kRes[LVL] = {
    16, 25, 41, 66, 107, 173, 279, 450,
    725, 1169, 1883, 3035, 4889, 7877, 12689, 20443};

__device__ __forceinline__ v16h mk16(v8h lo, v8h hi) {
    return __builtin_shufflevector(lo, hi, 0,1,2,3,4,5,6,7,8,9,10,11,12,13,14,15);
}

// A fragment (16x32 f16): lane<16 holds row M=(lane&15), K = ks*32 + {0..7, 16..23}
//                         lane>=16 holds same row,        K = ks*32 + {8..15, 24..31}
__device__ __forceinline__ v16h loadA(const _Float16* buf, int row, int ks, int lhi) {
    const _Float16* p = buf + row * ENC_K + ks * 32 + lhi * 8;
    v8h lo = *(const v8h*)p;
    v8h hi = *(const v8h*)(p + 16);
    return mk16(lo, hi);
}

// B fragment (32x16 f16) from transposed weights Wt[n][k] (row stride 64 halves):
// lane<16 holds col N=(lane&15), K = ks*32 + {0..7, 16..23} ; lane>=16 holds K = ks*32 + {8..15, 24..31}
__device__ __forceinline__ v16h loadB(const _Float16* wt, int col, int ks, int lhi) {
    const _Float16* p = wt + col * ENC_K + ks * 32 + lhi * 8;
    v8h lo = *(const v8h*)p;
    v8h hi = *(const v8h*)(p + 16);
    return mk16(lo, hi);
}

__global__ __launch_bounds__(BLK) void ngp_forward(
    const float* __restrict__ pos,     // [N,3]
    const float* __restrict__ dir,     // [N,3]
    const float* __restrict__ table,   // [16, 2^19, 2]
    const float* __restrict__ W0,      // [41,64]
    const float* __restrict__ W1,      // [64,64]
    const float* __restrict__ W2,      // [64,4]
    float* __restrict__ out,           // [N,4]
    int npts)
{
    __shared__ __align__(16) _Float16 sENC[BLK * ENC_K];   // 16KB (enc, reused as hidden)
    __shared__ __align__(16) _Float16 sW0T[64 * ENC_K];    // 8KB  W0^T padded K->64
    __shared__ __align__(16) _Float16 sW1T[64 * ENC_K];    // 8KB  W1^T
    __shared__ __align__(16) _Float16 sW2T[16 * ENC_K];    // 2KB  W2^T padded N->16
    __shared__ __align__(16) float    sOUT[BLK * 4];       // 2KB  raw layer-2 output

    const int tid = threadIdx.x;

    // ---- cooperative weight preload (transpose + f16 convert + zero pad) ----
    for (int i = tid; i < 64 * 64; i += BLK) {
        int n = i >> 6, k = i & 63;
        sW0T[n * ENC_K + k] = (_Float16)((k < 41) ? W0[k * 64 + n] : 0.f);
        sW1T[n * ENC_K + k] = (_Float16)(W1[k * 64 + n]);
    }
    for (int i = tid; i < 16 * 64; i += BLK) {
        int n = i >> 6, k = i & 63;
        sW2T[n * ENC_K + k] = (_Float16)((n < 4) ? W2[k * 4 + n] : 0.f);
    }
    __syncthreads();

    // Clamp (not early-return): keep EXEC all-ones for WMMA legality.
    int pt = blockIdx.x * BLK + tid;
    if (pt >= npts) pt = npts - 1;

    // ---- cube scene contraction ----
    float px = pos[3 * pt + 0], py = pos[3 * pt + 1], pz = pos[3 * pt + 2];
    float ax = fabsf(px), ay = fabsf(py), az = fabsf(pz);
    float m = fmaxf(ax, fmaxf(ay, az));
    // signed major component (argmax of |p|, first-match order x,y,z like argmax)
    float major = (ax >= ay && ax >= az) ? px : ((ay >= az) ? py : pz);
    bool  inner = (m <= 4.0f);
    float safe  = inner ? 1.0f : major;
    float nmaj  = (safe > 0.f) ? (2.0f - 4.0f / safe) : (-2.0f - 4.0f / safe);
    float ratio = nmaj / safe;
    float mx = inner ? (px * 0.25f) : (px * ratio);
    float my = inner ? (py * 0.25f) : (py * ratio);
    float mz = inner ? (pz * 0.25f) : (pz * ratio);
    mx = (mx + 2.0f) * 0.25f; my = (my + 2.0f) * 0.25f; mz = (mz + 2.0f) * 0.25f;

    // ---- multi-resolution hash grid encode -> sENC[row][0..31] as f16 ----
    _Float16* myrow = sENC + tid * ENC_K;
    const float2* tb2 = (const float2*)table;
#pragma unroll
    for (int l = 0; l < LVL; ++l) {
        const int res = kRes[l];
        const bool dense = ((long long)(res + 1) * (res + 1) * (res + 1) <= (long long)TBL);
        float fx = mx * (float)res, fy = my * (float)res, fz = mz * (float)res;
        float xf = floorf(fx), yf = floorf(fy), zf = floorf(fz);
        float wx = fx - xf, wy = fy - yf, wz = fz - zf;
        unsigned xi = (unsigned)(int)xf, yi = (unsigned)(int)yf, zi = (unsigned)(int)zf;
        const float2* tl = tb2 + (size_t)l * TBL;
        float f0 = 0.f, f1 = 0.f;
#pragma unroll
        for (int c = 0; c < 8; ++c) {
            unsigned dx = c & 1, dy = (c >> 1) & 1, dz = (c >> 2) & 1;
            unsigned cx = xi + dx, cy = yi + dy, cz = zi + dz;
            unsigned idx;
            if (dense) {
                idx = cx + cy * (unsigned)(res + 1) + cz * (unsigned)((res + 1) * (res + 1));
                if (idx >= TBL) idx = TBL - 1;       // boundary safety
            } else {
                idx = (cx * 1u) ^ (cy * 2654435761u) ^ (cz * 805459861u);
                idx &= (TBL - 1);
            }
            float w = (dx ? wx : 1.f - wx) * (dy ? wy : 1.f - wy) * (dz ? wz : 1.f - wz);
            float2 t = tl[idx];                       // global_load_b64, L2-resident
            f0 = fmaf(w, t.x, f0);
            f1 = fmaf(w, t.y, f1);
        }
        union { _Float16 h[2]; unsigned u; } pk;
        pk.h[0] = (_Float16)f0; pk.h[1] = (_Float16)f1;
        *(unsigned*)(myrow + 2 * l) = pk.u;           // ds_store_b32
    }

    // ---- spherical harmonics deg-3 -> sENC[row][32..40], zero pad [41..63] ----
    {
        float dxv = dir[3 * pt + 0], dyv = dir[3 * pt + 1], dzv = dir[3 * pt + 2];
        float s[9];
        s[0] = 0.28209479177387814f;
        s[1] = -0.4886025119029199f * dyv;
        s[2] =  0.4886025119029199f * dzv;
        s[3] = -0.4886025119029199f * dxv;
        s[4] =  1.0925484305920792f * dxv * dyv;
        s[5] = -1.0925484305920792f * dyv * dzv;
        s[6] =  0.31539156525252005f * (2.0f * dzv * dzv - dxv * dxv - dyv * dyv);
        s[7] = -1.0925484305920792f * dxv * dzv;
        s[8] =  0.5462742152960396f * (dxv * dxv - dyv * dyv);
#pragma unroll
        for (int i = 0; i < 4; ++i) {
            union { _Float16 h[2]; unsigned u; } pk;
            pk.h[0] = (_Float16)s[2 * i]; pk.h[1] = (_Float16)s[2 * i + 1];
            *(unsigned*)(myrow + 32 + 2 * i) = pk.u;
        }
        v8h z8 = {};
        *(v8h*)(myrow + 40) = z8;                     // zeros 40..47 (b128)
        *(v8h*)(myrow + 48) = z8;
        *(v8h*)(myrow + 56) = z8;
        myrow[40] = (_Float16)s[8];                   // overwrite slot 40 (in-order DS)
    }
    // NOTE: each wave consumes exactly the 32 rows its own lanes just wrote,
    // and DS ops are in-order within a wave -> no barrier needed here.

    const int lane = tid & 31;
    const int lrow = lane & 15;
    const int lhi  = lane >> 4;
    const int rowbase = (tid >> 5) * 32;

    // ---- layer 0 and layer 1: [32 x 64] @ [64 x 64], ReLU, in-place in sENC ----
#pragma unroll
    for (int layer = 0; layer < 2; ++layer) {
        const _Float16* wt = (layer == 0) ? sW0T : sW1T;
        v8f acc[2][4] = {};
#pragma unroll
        for (int ks = 0; ks < 2; ++ks) {
            v16h a0 = loadA(sENC, rowbase + lrow,      ks, lhi);
            v16h a1 = loadA(sENC, rowbase + 16 + lrow, ks, lhi);
#pragma unroll
            for (int nt = 0; nt < 4; ++nt) {
                v16h b = loadB(wt, nt * 16 + lrow, ks, lhi);
                acc[0][nt] = __builtin_amdgcn_wmma_f32_16x16x32_f16(
                    false, a0, false, b, (short)0, acc[0][nt], false, false);
                acc[1][nt] = __builtin_amdgcn_wmma_f32_16x16x32_f16(
                    false, a1, false, b, (short)0, acc[1][nt], false, false);
            }
        }
        // ReLU + write back as f16 (D layout: VGPR j -> M = j + 8*lhi, N = lrow)
#pragma unroll
        for (int mt = 0; mt < 2; ++mt)
#pragma unroll
            for (int nt = 0; nt < 4; ++nt)
#pragma unroll
                for (int j = 0; j < 8; ++j) {
                    float v = acc[mt][nt][j];
                    v = v > 0.f ? v : 0.f;
                    sENC[(rowbase + mt * 16 + j + lhi * 8) * ENC_K + nt * 16 + lrow] =
                        (_Float16)v;
                }
    }

    // ---- layer 2: [32 x 64] @ [64 x 16] (cols 4..15 are zero pad) ----
    {
        v8f acc[2] = {};
#pragma unroll
        for (int ks = 0; ks < 2; ++ks) {
            v16h a0 = loadA(sENC, rowbase + lrow,      ks, lhi);
            v16h a1 = loadA(sENC, rowbase + 16 + lrow, ks, lhi);
            v16h b  = loadB(sW2T, lrow, ks, lhi);
            acc[0] = __builtin_amdgcn_wmma_f32_16x16x32_f16(
                false, a0, false, b, (short)0, acc[0], false, false);
            acc[1] = __builtin_amdgcn_wmma_f32_16x16x32_f16(
                false, a1, false, b, (short)0, acc[1], false, false);
        }
        if (lrow < 4) {                               // only real output columns
#pragma unroll
            for (int mt = 0; mt < 2; ++mt)
#pragma unroll
                for (int j = 0; j < 8; ++j)
                    sOUT[(rowbase + mt * 16 + j + lhi * 8) * 4 + lrow] = acc[mt][j];
        }
    }

    // ---- per-point nonlinearities (wave-local rows; DS in-order, no barrier) ----
    float4 o = *(const float4*)(sOUT + tid * 4);
    float raw = o.x;
    float density = fminf(__expf(raw), fmaxf(raw, 0.f) + 1000000.0f);
    float r = 1.0f / (1.0f + __expf(-o.y));
    float g = 1.0f / (1.0f + __expf(-o.z));
    float b = 1.0f / (1.0f + __expf(-o.w));
    float4 res4 = make_float4(density, r, g, b);
    *(float4*)(out + (size_t)pt * 4) = res4;          // global_store_b128
}

extern "C" void kernel_launch(void* const* d_in, const int* in_sizes, int n_in,
                              void* d_out, int out_size, void* d_ws, size_t ws_size,
                              hipStream_t stream) {
    (void)n_in; (void)d_ws; (void)ws_size; (void)out_size;
    const float* pos = (const float*)d_in[0];
    const float* dir = (const float*)d_in[1];
    const float* tbl = (const float*)d_in[2];
    const float* W0  = (const float*)d_in[3];
    const float* W1  = (const float*)d_in[4];
    const float* W2  = (const float*)d_in[5];
    float* out = (float*)d_out;
    int npts = in_sizes[0] / 3;                       // 65536*32 = 2,097,152
    int nblk = (npts + BLK - 1) / BLK;
    ngp_forward<<<nblk, BLK, 0, stream>>>(pos, dir, tbl, W0, W1, W2, out, npts);
}
